// DivTree_14216341750023
// MI455X (gfx1250) — matrix operand order repaired and checked by reading, not compiled, not run on hardware
//
#include <hip/hip_runtime.h>

// Fused divtree-MoE forward for MI455X (gfx1250, wave32, WMMA + async-to-LDS).
//
// out[b,a,:] = relu(x[b,a,:] @ W1[r(a)][:H] + b1_eff) @ W2[r(a)] + b2[r(a)]
//   with b1_eff = b1[r(a)] + W1[r(a)][H+a, :]   (one-hot fold)
//
// Compute-bound in f32 (AI ~ 400 FLOP/B) -> bf16 WMMA w/ f32 accumulation.
// One workgroup per WGP (~311 KB LDS) -> W1 chunks double-buffered through an
// LDS f32 staging buffer filled by GLOBAL_LOAD_ASYNC_TO_LDS (ASYNCcnt),
// overlapping the HBM copy of chunk nc+1 with the WMMAs of chunk nc.
// All LDS tiles use padded row strides (520 / 40 halfwords) so fragment
// loads hit all 64 banks instead of a 16-way same-bank conflict.

#define B_SZ   2048
#define A_SZ   50
#define H_SZ   512
#define F_SZ   562
#define NACT   64
#define BM     128      // batch rows per workgroup
#define HCHUNK 32       // H columns per chunk
#define NCHUNK (H_SZ / HCHUNK)   // 16
#define XS     520      // padded row stride (halfwords) for 512-wide tiles
#define HS     40       // padded row stride (halfwords) for 32-wide h tile

typedef __attribute__((ext_vector_type(16))) __bf16 v16bf;
typedef __attribute__((ext_vector_type(8)))  float  v8f;
typedef unsigned short u16;
typedef int v4i __attribute__((vector_size(16)));
typedef __attribute__((address_space(1))) v4i* gv4i_p;   // global (AS1)
typedef __attribute__((address_space(3))) v4i* lv4i_p;   // LDS (AS3)

struct Frag { union { v16bf v; uint4 q[2]; }; };

__device__ __forceinline__ u16 f32_to_bf16(float f) {
  unsigned int u = __float_as_uint(f);
  u += 0x7FFFu + ((u >> 16) & 1u);          // round to nearest even
  return (u16)(u >> 16);
}

// A operand (16x32 bf16, MxK), row-major source in LDS (ISA 7.12.2):
// lanes 0-15: v0..3 = K k0..k0+7, v4..7 = K k0+16..k0+23 ; lanes 16-31: +8.
__device__ __forceinline__ Frag load_frag_a(const u16* base, int row, int stride,
                                            int k0, int lane) {
  Frag f;
  const u16* p = base + (row + (lane & 15)) * stride + k0 + ((lane >> 4) << 3);
  f.q[0] = *(const uint4*)(p);
  f.q[1] = *(const uint4*)(p + 16);
  return f;
}

// B operand (32x16 bf16, KxN), stored N-major ([n][k]) in LDS:
// lanes 0-15: v0..3 = K k0..k0+7, v4..7 = K k0+16..k0+23 ; lanes 16-31: +8.
__device__ __forceinline__ Frag load_frag_b(const u16* base, int col, int stride,
                                            int k0, int lane) {
  Frag f;
  const u16* p = base + (col + (lane & 15)) * stride + k0 + ((lane >> 4) << 3);
  f.q[0] = *(const uint4*)(p);
  f.q[1] = *(const uint4*)(p + 16);
  return f;
}

__device__ __forceinline__ v8f wmma_bf16(const Frag& a, const Frag& b, v8f c) {
  return __builtin_amdgcn_wmma_f32_16x16x32_bf16(
      /*neg_a=*/false, a.v, /*neg_b=*/false, b.v,
      /*c_mod=*/(short)0, c, /*reuse_a=*/false, /*reuse_b=*/false);
}

// Async DMA: 16 bytes global -> LDS per lane, tracked by ASYNCcnt.
__device__ __forceinline__ void async_copy_b128(const void* g, void* l) {
  __builtin_amdgcn_global_load_async_to_lds_b128(
      (gv4i_p)g, (lv4i_p)l, /*offset=*/0, /*cpol=*/0);
}

__device__ __forceinline__ void wait_asynccnt0() {
  asm volatile("s_wait_asynccnt 0x0" ::: "memory");
}

__global__ __launch_bounds__(256, 1)
void moe_divtree_fused(const float* __restrict__ x,
                       const float* __restrict__ W1,
                       const float* __restrict__ b1,
                       const float* __restrict__ W2,
                       const float* __restrict__ b2,
                       const int*  __restrict__ routing,
                       float* __restrict__ out) {
  extern __shared__ char smem[];
  float* lds_stg = (float*)smem;                    // [H][HCHUNK] f32     64 KB
  u16*   lds_x   = (u16*)(lds_stg + H_SZ * HCHUNK); // [BM][XS]    bf16   130 KB
  u16*   lds_w2  = lds_x  + BM * XS;                // [NACT][XS]  bf16    65 KB
  u16*   lds_w1  = lds_w2 + NACT * XS;              // [HCHUNK][XS] bf16   32.5 KB
  u16*   lds_h   = lds_w1 + HCHUNK * XS;            // [BM][HS]    bf16    10 KB
  float* lds_b1  = (float*)(lds_h + BM * HS);       // [H]         f32      2 KB

  const int tid  = threadIdx.x;
  const int lane = tid & 31;
  const int wave = tid >> 5;
  const int a    = blockIdx.x % A_SZ;
  const int m0   = (blockIdx.x / A_SZ) * BM;
  const int ar   = routing[a];

  const float* w1p = W1 + (size_t)ar * F_SZ * H_SZ;
  const float* w2p = W2 + (size_t)ar * H_SZ * NACT;
  const float* b1p = b1 + (size_t)ar * H_SZ;
  const float* b2p = b2 + (size_t)ar * NACT;

  // --- issue async DMA of W1 chunk 0 (f32, raw) into the staging buffer;
  //     it completes under the direct staging of x/W2 below. -------------
  // chunk layout in stg: [k][c] (k=0..511 rows of HCHUNK f32, unpadded)
  {
    const int nc = 0;
#pragma unroll
    for (int it = 0; it < (H_SZ * HCHUNK) / (256 * 4); ++it) {  // 16 issues
      int ef = (it * 256 + tid) * 4;        // f32 element index
      int k = ef >> 5, c = ef & (HCHUNK - 1);
      async_copy_b128(w1p + (size_t)k * H_SZ + nc * HCHUNK + c, lds_stg + ef);
    }
  }

  // ---- stage x tile f32 -> bf16, row-major [m][k], padded stride XS ----
  for (int it = 0; it < (BM * H_SZ) / (256 * 4); ++it) {
    int e = (it * 256 + tid) * 4;
    int m = e >> 9, k = e & (H_SZ - 1);
    float4 f = *(const float4*)(x + ((size_t)(m0 + m) * A_SZ + a) * H_SZ + k);
    unsigned int u0 = f32_to_bf16(f.x) | ((unsigned int)f32_to_bf16(f.y) << 16);
    unsigned int u1 = f32_to_bf16(f.z) | ((unsigned int)f32_to_bf16(f.w) << 16);
    *(uint2*)(lds_x + m * XS + k) = make_uint2(u0, u1);
  }
  // ---- stage W2 f32 -> bf16, transposed [n][k], padded stride XS ----
  for (int it = 0; it < (H_SZ * NACT) / (256 * 4); ++it) {
    int e = (it * 256 + tid) * 4;
    int k = e >> 6, n = e & (NACT - 1);
    float4 f = *(const float4*)(w2p + (size_t)k * NACT + n);
    lds_w2[(n + 0) * XS + k] = f32_to_bf16(f.x);
    lds_w2[(n + 1) * XS + k] = f32_to_bf16(f.y);
    lds_w2[(n + 2) * XS + k] = f32_to_bf16(f.z);
    lds_w2[(n + 3) * XS + k] = f32_to_bf16(f.w);
  }
  // ---- effective bias (folds the one-hot block of the concat) ----
  for (int i = tid; i < H_SZ; i += 256)
    lds_b1[i] = b1p[i] + w1p[(size_t)(H_SZ + a) * H_SZ + i];

  // ---- chunk 0: wait for DMA, convert stg(f32) -> lds_w1(bf16, [nn][k]) ----
  wait_asynccnt0();
  __syncthreads();
  for (int it = 0; it < (H_SZ * HCHUNK) / (256 * 4); ++it) {
    int ef = (it * 256 + tid) * 4;
    int k = ef >> 5, c = ef & (HCHUNK - 1);
    float4 f = *(const float4*)(lds_stg + ef);
    lds_w1[(c + 0) * XS + k] = f32_to_bf16(f.x);
    lds_w1[(c + 1) * XS + k] = f32_to_bf16(f.y);
    lds_w1[(c + 2) * XS + k] = f32_to_bf16(f.z);
    lds_w1[(c + 3) * XS + k] = f32_to_bf16(f.w);
  }
  __syncthreads();

  const int mbase = wave * 16;              // this wave's 16 batch rows
  const int ncol = lane & 15, hi = lane >> 4;
  const v8f vz = {0.f, 0.f, 0.f, 0.f, 0.f, 0.f, 0.f, 0.f};
  v8f oacc[4];
#pragma unroll
  for (int nt = 0; nt < 4; ++nt) oacc[nt] = vz;

  for (int nc = 0; nc < NCHUNK; ++nc) {
    // ---- kick off DMA of chunk nc+1 while we compute on chunk nc ----
    if (nc + 1 < NCHUNK) {
#pragma unroll
      for (int it = 0; it < (H_SZ * HCHUNK) / (256 * 4); ++it) {
        int ef = (it * 256 + tid) * 4;
        int k = ef >> 5, c = ef & (HCHUNK - 1);
        async_copy_b128(w1p + (size_t)k * H_SZ + (nc + 1) * HCHUNK + c,
                        lds_stg + ef);
      }
    }

    // ---- GEMM1: (16 x 512) @ (512 x 32) -> h chunk, WMMA bf16 ----
    v8f acc1[2];
#pragma unroll
    for (int nt = 0; nt < 2; ++nt) acc1[nt] = vz;
#pragma unroll 4
    for (int k0 = 0; k0 < H_SZ; k0 += 32) {
      Frag af = load_frag_a(lds_x, mbase, XS, k0, lane);
#pragma unroll
      for (int nt = 0; nt < 2; ++nt) {
        Frag bf = load_frag_b(lds_w1, nt * 16, XS, k0, lane);
        acc1[nt] = wmma_bf16(af, bf, acc1[nt]);
      }
    }
    // bias + relu -> bf16 h chunk in LDS (row-major [m][kk], stride HS)
#pragma unroll
    for (int nt = 0; nt < 2; ++nt) {
      float bias = lds_b1[nc * HCHUNK + nt * 16 + ncol];
#pragma unroll
      for (int i = 0; i < 8; ++i) {
        float v = acc1[nt][i] + bias;
        v = v > 0.f ? v : 0.f;
        lds_h[(mbase + hi * 8 + i) * HS + nt * 16 + ncol] = f32_to_bf16(v);
      }
    }
    // ---- GEMM2 partial: oacc += h_chunk @ W2[nc*32 : nc*32+32, :] ----
    // (h rows written and read by the same wave -> DScnt ordering suffices)
    {
      Frag af = load_frag_a(lds_h, mbase, HS, 0, lane);
#pragma unroll
      for (int nt = 0; nt < 4; ++nt) {
        Frag bf = load_frag_b(lds_w2, nt * 16, XS, nc * HCHUNK, lane);
        oacc[nt] = wmma_bf16(af, bf, oacc[nt]);
      }
    }

    // ---- rotate: drain DMA, convert stg -> lds_w1 for chunk nc+1 ----
    if (nc + 1 < NCHUNK) {
      wait_asynccnt0();
      __syncthreads();   // all GEMM1 readers of lds_w1 done; stg data landed
      for (int it = 0; it < (H_SZ * HCHUNK) / (256 * 4); ++it) {
        int ef = (it * 256 + tid) * 4;
        int k = ef >> 5, c = ef & (HCHUNK - 1);
        float4 f = *(const float4*)(lds_stg + ef);
        lds_w1[(c + 0) * XS + k] = f32_to_bf16(f.x);
        lds_w1[(c + 1) * XS + k] = f32_to_bf16(f.y);
        lds_w1[(c + 2) * XS + k] = f32_to_bf16(f.z);
        lds_w1[(c + 3) * XS + k] = f32_to_bf16(f.w);
      }
      __syncthreads();
    }
  }

  // ---- epilogue: + b2, f32 store to out (B, A, NACT); lanes 0-15 cover
  // N=0..15 at consecutive addresses -> coalesced global_store_b32 ----
#pragma unroll
  for (int nt = 0; nt < 4; ++nt) {
    float bb = b2p[nt * 16 + ncol];
#pragma unroll
    for (int i = 0; i < 8; ++i) {
      int m = m0 + mbase + hi * 8 + i;
      out[((size_t)m * A_SZ + a) * NACT + nt * 16 + ncol] = oacc[nt][i] + bb;
    }
  }
}

extern "C" void kernel_launch(void* const* d_in, const int* in_sizes, int n_in,
                              void* d_out, int out_size, void* d_ws, size_t ws_size,
                              hipStream_t stream) {
  const float* x   = (const float*)d_in[0];
  const float* W1  = (const float*)d_in[1];
  const float* b1  = (const float*)d_in[2];
  const float* W2  = (const float*)d_in[3];
  const float* b2  = (const float*)d_in[4];
  const int*   rt  = (const int*)d_in[5];
  float*       out = (float*)d_out;

  constexpr size_t SMEM =
      (size_t)H_SZ * HCHUNK * sizeof(float)                  // stg     65536
    + (size_t)(BM * XS + NACT * XS + HCHUNK * XS + BM * HS)
        * sizeof(u16)                                        // bf16 tiles
    + (size_t)H_SZ * sizeof(float);                          // b1_eff   2048
  // = 310,784 B of the WGP's 320 KB

  (void)in_sizes; (void)n_in; (void)out_size; (void)d_ws; (void)ws_size;
  (void)hipFuncSetAttribute((const void*)moe_divtree_fused,
                            hipFuncAttributeMaxDynamicSharedMemorySize,
                            (int)SMEM);

  const int nblk = (B_SZ / BM) * A_SZ;   // 16 * 50 = 800 workgroups
  moe_divtree_fused<<<dim3(nblk), dim3(256), SMEM, stream>>>(
      x, W1, b1, W2, b2, rt, out);
}
